// HetGraphModel_8160437862810
// MI455X (gfx1250) — matrix operand order repaired and checked by reading, not compiled, not run on hardware
//
#include <hip/hip_runtime.h>

// Problem constants (match reference)
#define Nn 100000
#define Ee 500000
#define Rr 3
#define Ll 2
#define Ff 300
#define Hh 256
#define Cc 23

typedef __attribute__((ext_vector_type(16))) __bf16 v16bf;
typedef __attribute__((ext_vector_type(8)))  __bf16 v8bf;
typedef __attribute__((ext_vector_type(8)))  float  v8f;

union Frag16 { v16bf v; v8bf h[2]; };

__device__ __forceinline__ void bf_split(float x, __bf16* hi, __bf16* lo) {
  __bf16 h = (__bf16)x;          // RNE to bf16
  *hi = h;
  *lo = (__bf16)(x - (float)h);  // residual in bf16
}

// ---------------------------------------------------------------------------
// Staging: load a 64x32 A slice (f32 -> bf16 hi/lo, optional per-row scale)
// and a 32x64 B slice (stored transposed [n][k]) into LDS.
// TAIL=false is the hot path: unconditional float4 loads only.
// ---------------------------------------------------------------------------
template <bool TAIL, bool SCALE>
__device__ __forceinline__ void stage_tiles(
    const float* __restrict__ A, const float* __restrict__ W,
    const float* __restrict__ rowscale,
    __bf16* __restrict__ Ahi, __bf16* __restrict__ Alo,
    __bf16* __restrict__ Bhi, __bf16* __restrict__ Blo,
    int tid, int rowBase, int colBase, int k0,
    int M, int K, int lda, int ncols)
{
  // A tile: 64 rows x 32 k
  #pragma unroll
  for (int j = 0; j < 2; ++j) {
    int lin = tid + j * 256;
    int row = lin >> 3;
    int kq  = (lin & 7) * 4;
    float t4[4] = {0.f, 0.f, 0.f, 0.f};
    int grow = rowBase + row;
    if (grow < M) {
      const float* ap = A + (size_t)grow * lda + k0 + kq;
      if (!TAIL || (k0 + kq + 3 < K)) {
        float4 vv = *(const float4*)ap;
        t4[0] = vv.x; t4[1] = vv.y; t4[2] = vv.z; t4[3] = vv.w;
      } else {
        #pragma unroll
        for (int u = 0; u < 4; ++u) if (k0 + kq + u < K) t4[u] = ap[u];
      }
      if (SCALE) {
        float s = rowscale[grow];
        t4[0] *= s; t4[1] *= s; t4[2] *= s; t4[3] *= s;
      }
    }
    #pragma unroll
    for (int u = 0; u < 4; ++u)
      bf_split(t4[u], &Ahi[row * 32 + kq + u], &Alo[row * 32 + kq + u]);
  }
  // B tile: 32 k x 64 n, transposed into [n][k]
  #pragma unroll
  for (int j = 0; j < 2; ++j) {
    int lin = tid + j * 256;
    int k   = lin >> 4;
    int nq  = (lin & 15) * 4;
    float t4[4] = {0.f, 0.f, 0.f, 0.f};
    if (!TAIL || (k0 + k < K)) {
      float4 vv = *(const float4*)(W + (size_t)(k0 + k) * ncols + colBase + nq);
      t4[0] = vv.x; t4[1] = vv.y; t4[2] = vv.z; t4[3] = vv.w;
    }
    #pragma unroll
    for (int u = 0; u < 4; ++u)
      bf_split(t4[u], &Bhi[(nq + u) * 32 + k], &Blo[(nq + u) * 32 + k]);
  }
}

// ---------------------------------------------------------------------------
// Compute: per-wave, load A fragment (16x32) + two B fragments (32x16) from
// LDS (contiguous ds_load_b128 per the gfx1250 16-bit WMMA lane layouts) and
// issue 3 split-bf16 WMMAs per output tile.
// ---------------------------------------------------------------------------
__device__ __forceinline__ void compute_tiles(
    const __bf16* __restrict__ Ahi, const __bf16* __restrict__ Alo,
    const __bf16* __restrict__ Bhi, const __bf16* __restrict__ Blo,
    int lane, int tm, int tn0, v8f acc[2])
{
  Frag16 ahi, alo;
  {
    int m = lane & 15, kb = (lane >> 4) * 8;
    const __bf16* p = &Ahi[(tm * 16 + m) * 32 + kb];
    ahi.h[0] = *(const v8bf*)p;
    ahi.h[1] = *(const v8bf*)(p + 16);
    const __bf16* q = &Alo[(tm * 16 + m) * 32 + kb];
    alo.h[0] = *(const v8bf*)q;
    alo.h[1] = *(const v8bf*)(q + 16);
  }
  #pragma unroll
  for (int t = 0; t < 2; ++t) {
    int tn = tn0 + t;
    int n = lane & 15, kb2 = (lane >> 4) * 8;
    Frag16 bhi, blo;
    const __bf16* p = &Bhi[(tn * 16 + n) * 32 + kb2];
    bhi.h[0] = *(const v8bf*)p;
    bhi.h[1] = *(const v8bf*)(p + 16);
    const __bf16* q = &Blo[(tn * 16 + n) * 32 + kb2];
    blo.h[0] = *(const v8bf*)q;
    blo.h[1] = *(const v8bf*)(q + 16);
    acc[t] = __builtin_amdgcn_wmma_f32_16x16x32_bf16(false, ahi.v, false, bhi.v, (short)0, acc[t], false, false);
    acc[t] = __builtin_amdgcn_wmma_f32_16x16x32_bf16(false, ahi.v, false, blo.v, (short)0, acc[t], false, false);
    acc[t] = __builtin_amdgcn_wmma_f32_16x16x32_bf16(false, alo.v, false, bhi.v, (short)0, acc[t], false, false);
  }
}

// ---------------------------------------------------------------------------
// Split-bf16 (3-term) WMMA GEMM with double-buffered LDS:
//   C[M x ncols] = (rowscale ? diag(rowscale) : I) * A[M x K] * W[K x ncols]
//                  + bias (+ C if accum)
// Block tile 64x64, 256 threads = 8 waves. Iteration s stages tile s+1 while
// computing tile s, one barrier per k-step.
// ---------------------------------------------------------------------------
__global__ __launch_bounds__(256) void gemm_bf16x3(
    const float* __restrict__ A, const float* __restrict__ W,
    const float* __restrict__ bias, const float* __restrict__ rowscale,
    float* __restrict__ C, int M, int K, int lda, int ncols, int accum)
{
  __shared__ __bf16 AhiS[2][64 * 32], AloS[2][64 * 32];
  __shared__ __bf16 BhiS[2][64 * 32], BloS[2][64 * 32];   // 32 KB total

  const int tid  = threadIdx.x;
  const int lane = tid & 31;
  const int wave = tid >> 5;
  const int tm   = wave & 3;
  const int tn0  = (wave >> 2) * 2;
  const int rowBase = blockIdx.x * 64;
  const int colBase = blockIdx.y * 64;
  const bool scale = (rowscale != nullptr);

  v8f acc[2] = {};

  const int nsteps = (K + 31) >> 5;

  auto stageStep = [&](int buf, int s) {
    int k0 = s << 5;
    if (k0 + 32 <= K) {   // uniform branch: hot path has no per-element guards
      if (scale) stage_tiles<false, true >(A, W, rowscale, AhiS[buf], AloS[buf], BhiS[buf], BloS[buf], tid, rowBase, colBase, k0, M, K, lda, ncols);
      else       stage_tiles<false, false>(A, W, rowscale, AhiS[buf], AloS[buf], BhiS[buf], BloS[buf], tid, rowBase, colBase, k0, M, K, lda, ncols);
    } else {
      if (scale) stage_tiles<true,  true >(A, W, rowscale, AhiS[buf], AloS[buf], BhiS[buf], BloS[buf], tid, rowBase, colBase, k0, M, K, lda, ncols);
      else       stage_tiles<true,  false>(A, W, rowscale, AhiS[buf], AloS[buf], BhiS[buf], BloS[buf], tid, rowBase, colBase, k0, M, K, lda, ncols);
    }
  };

  stageStep(0, 0);
  __syncthreads();
  int cur = 0;
  for (int s = 1; s < nsteps; ++s) {
    stageStep(cur ^ 1, s);    // prefetch next tile into alternate buffer
    compute_tiles(AhiS[cur], AloS[cur], BhiS[cur], BloS[cur], lane, tm, tn0, acc);
    __syncthreads();
    cur ^= 1;
  }
  compute_tiles(AhiS[cur], AloS[cur], BhiS[cur], BloS[cur], lane, tm, tn0, acc);

  // epilogue: C/D layout lanes 0-15 -> M=v, lanes 16-31 -> M=v+8
  int n = lane & 15, mb = (lane >> 4) * 8;
  #pragma unroll
  for (int t = 0; t < 2; ++t) {
    int col = colBase + (tn0 + t) * 16 + n;
    float bsv = bias ? bias[col] : 0.f;
    #pragma unroll
    for (int v = 0; v < 8; ++v) {
      int row = rowBase + tm * 16 + mb + v;
      if (row < M) {
        size_t off = (size_t)row * ncols + col;
        float val = acc[t][v] + bsv;
        if (accum) val += C[off];
        C[off] = val;
      }
    }
  }
}

// ---------------------------------------------------------------------------
// Degree counting and rsqrt(clip(deg,1)) finalize
// ---------------------------------------------------------------------------
__global__ __launch_bounds__(256) void degree_count(
    const int* __restrict__ src, const int* __restrict__ dst,
    float* __restrict__ cnt_out, float* __restrict__ cnt_in)
{
  int e = blockIdx.x * 256 + threadIdx.x;
  if (e >= Rr * Ee) return;
  int r = e / Ee;
  atomicAdd(&cnt_out[(size_t)r * Nn + src[e]], 1.f);
  atomicAdd(&cnt_in [(size_t)r * Nn + dst[e]], 1.f);
}

__global__ __launch_bounds__(256) void finalize_deg(float* __restrict__ d, int n)
{
  int i = blockIdx.x * 256 + threadIdx.x;
  if (i >= n) return;
  d[i] = rsqrtf(fmaxf(d[i], 1.f));
}

// ---------------------------------------------------------------------------
// Edge gather-scatter: m[dst] += h[src] * isq_out[src]   (64 lanes / edge)
// ---------------------------------------------------------------------------
__global__ __launch_bounds__(256) void scatter_edges(
    const float* __restrict__ h, const int* __restrict__ src,
    const int* __restrict__ dst, const float* __restrict__ isq,
    float* __restrict__ m)
{
  int gid = blockIdx.x * 256 + threadIdx.x;
  int e = gid >> 6;
  int c = (gid & 63) * 4;
  if (e >= Ee) return;
  int s = src[e], d = dst[e];
  // warm L2 for a future edge's source row (global_prefetch_b8)
  int e2 = e + 16384;
  if (e2 < Ee) __builtin_prefetch(h + (size_t)src[e2] * Hh + c, 0, 0);
  float sc = isq[s];
  float4 v = *(const float4*)(h + (size_t)s * Hh + c);
  float* mp = m + (size_t)d * Hh + c;
  atomicAdd(mp + 0, v.x * sc);
  atomicAdd(mp + 1, v.y * sc);
  atomicAdd(mp + 2, v.z * sc);
  atomicAdd(mp + 3, v.w * sc);
}

// ---------------------------------------------------------------------------
// BatchNorm plumbing
// ---------------------------------------------------------------------------
__global__ __launch_bounds__(256) void col_stats(
    const float* __restrict__ h, float* __restrict__ stats, int nrows)
{
  int col = threadIdx.x;
  int r0 = blockIdx.x * 128;
  int rend = min(r0 + 128, nrows);
  float s = 0.f, s2 = 0.f;
  for (int r = r0; r < rend; ++r) {
    float v = h[(size_t)r * Hh + col];
    s += v; s2 += v * v;
  }
  atomicAdd(&stats[col], s);
  atomicAdd(&stats[Hh + col], s2);
}

__global__ __launch_bounds__(256) void combine_stats(
    const float* __restrict__ hsum, const float* __restrict__ hres, float invR,
    float* __restrict__ hcomb, float* __restrict__ stats, int nrows)
{
  int col = threadIdx.x;
  int r0 = blockIdx.x * 128;
  int rend = min(r0 + 128, nrows);
  float s = 0.f, s2 = 0.f;
  for (int r = r0; r < rend; ++r) {
    size_t off = (size_t)r * Hh + col;
    float v = hsum[off] * invR + hres[off];
    hcomb[off] = v;
    s += v; s2 += v * v;
  }
  atomicAdd(&stats[col], s);
  atomicAdd(&stats[Hh + col], s2);
}

__global__ void bn_finalize(const float* __restrict__ stats,
                            const float* __restrict__ g, const float* __restrict__ b,
                            float inv_n, float* __restrict__ ss)
{
  int c = threadIdx.x;
  float mean = stats[c] * inv_n;
  float var  = stats[Hh + c] * inv_n - mean * mean;   // biased var
  float a = g[c] * rsqrtf(var + 1e-5f);
  ss[c] = a;
  ss[Hh + c] = b[c] - mean * a;
}

__global__ __launch_bounds__(256) void bn_apply_act(
    const float* __restrict__ x, const float* __restrict__ ss,
    float* __restrict__ y, int nrows, float slope)
{
  size_t i4 = (size_t)blockIdx.x * 256 + threadIdx.x;
  size_t tot = (size_t)nrows * (Hh / 4);
  if (i4 >= tot) return;
  int c = (int)(i4 & (Hh / 4 - 1)) * 4;
  float4 v = ((const float4*)x)[i4];
  float o[4] = {v.x, v.y, v.z, v.w};
  #pragma unroll
  for (int u = 0; u < 4; ++u) {
    float t = o[u] * ss[c + u] + ss[Hh + c + u];
    o[u] = t > 0.f ? t : t * slope;   // slope 0 => ReLU, 0.01 => LeakyReLU
  }
  float4 w; w.x = o[0]; w.y = o[1]; w.z = o[2]; w.w = o[3];
  ((float4*)y)[i4] = w;
}

// ---------------------------------------------------------------------------
// Tiny head GEMM: out[N x 23] = h[N x 256] @ W[256 x 23] + b, W staged in LDS
// ---------------------------------------------------------------------------
__global__ __launch_bounds__(256) void head_gemm(
    const float* __restrict__ h, const float* __restrict__ W,
    const float* __restrict__ b, float* __restrict__ out, int nrows)
{
  __shared__ float Ws[Hh * Cc];
  __shared__ float bs[Cc];
  for (int i = threadIdx.x; i < Hh * Cc; i += 256) Ws[i] = W[i];
  if (threadIdx.x < Cc) bs[threadIdx.x] = b[threadIdx.x];
  __syncthreads();
  int row = blockIdx.x * 256 + threadIdx.x;
  if (row >= nrows) return;
  float acc[Cc];
  #pragma unroll
  for (int c = 0; c < Cc; ++c) acc[c] = bs[c];
  const float* hp = h + (size_t)row * Hh;
  for (int k = 0; k < Hh; ++k) {
    float hv = hp[k];
    #pragma unroll
    for (int c = 0; c < Cc; ++c) acc[c] += hv * Ws[k * Cc + c];
  }
  float* op = out + (size_t)row * Cc;
  #pragma unroll
  for (int c = 0; c < Cc; ++c) op[c] = acc[c];
}

// ---------------------------------------------------------------------------
// Orchestration. Workspace layout (f32):
//   h[N*H] | hres[N*H] | bufB[N*H] | bufC[N*H] | isq[2*R*N] | stats[512] | ss[512]
// Total ~413 MB.
// ---------------------------------------------------------------------------
extern "C" void kernel_launch(void* const* d_in, const int* in_sizes, int n_in,
                              void* d_out, int out_size, void* d_ws, size_t ws_size,
                              hipStream_t stream)
{
  const float* x         = (const float*)d_in[0];
  const int*   edge_src  = (const int*)  d_in[1];
  const int*   edge_dst  = (const int*)  d_in[2];
  const float* W_feat    = (const float*)d_in[3];
  const float* b_feat    = (const float*)d_in[4];
  const float* g_feat    = (const float*)d_in[5];
  const float* beta_feat = (const float*)d_in[6];
  const float* gcn_W     = (const float*)d_in[7];
  const float* gcn_b     = (const float*)d_in[8];
  const float* skip_W    = (const float*)d_in[9];
  const float* skip_b    = (const float*)d_in[10];
  const float* bn_g      = (const float*)d_in[11];
  const float* bn_b      = (const float*)d_in[12];
  const float* W_c1      = (const float*)d_in[13];
  const float* b_c1      = (const float*)d_in[14];
  const float* g_c       = (const float*)d_in[15];
  const float* beta_c    = (const float*)d_in[16];
  const float* W_c2      = (const float*)d_in[17];
  const float* b_c2      = (const float*)d_in[18];
  float* out = (float*)d_out;

  const size_t NH = (size_t)Nn * Hh;
  float* h     = (float*)d_ws;
  float* hres  = h + NH;
  float* bufB  = hres + NH;          // scatter target / BN input scratch
  float* bufC  = bufB + NH;          // relation-sum / head scratch
  float* isq   = bufC + NH;          // [R*N] rsqrt(dout) then [R*N] rsqrt(din)
  float* stats = isq + (size_t)2 * Rr * Nn;
  float* ss    = stats + 2 * Hh;

  const float invN = 1.f / (float)Nn;
  dim3 blk(256);
  dim3 gemmGrid((Nn + 63) / 64, Hh / 64);
  int applyGrid = (int)((NH / 4 + 255) / 256);
  int statGrid  = (Nn + 127) / 128;

  // --- degrees (static across layers): counts -> rsqrt(clip(.,1)) in place
  hipMemsetAsync(isq, 0, sizeof(float) * 2 * Rr * Nn, stream);
  degree_count<<<(Rr * Ee + 255) / 256, blk, 0, stream>>>(
      edge_src, edge_dst, isq, isq + (size_t)Rr * Nn);
  finalize_deg<<<(2 * Rr * Nn + 255) / 256, blk, 0, stream>>>(isq, 2 * Rr * Nn);

  // --- feat_reduce: Linear + BN + ReLU
  gemm_bf16x3<<<gemmGrid, blk, 0, stream>>>(x, W_feat, b_feat, nullptr,
                                            bufB, Nn, Ff, Ff, Hh, 0);
  hipMemsetAsync(stats, 0, sizeof(float) * 2 * Hh, stream);
  col_stats<<<statGrid, blk, 0, stream>>>(bufB, stats, Nn);
  bn_finalize<<<1, blk, 0, stream>>>(stats, g_feat, beta_feat, invN, ss);
  bn_apply_act<<<applyGrid, blk, 0, stream>>>(bufB, ss, h, Nn, 0.f);

  // --- layers
  for (int l = 0; l < Ll; ++l) {
    gemm_bf16x3<<<gemmGrid, blk, 0, stream>>>(
        h, skip_W + (size_t)l * Hh * Hh, skip_b + (size_t)l * Hh, nullptr,
        hres, Nn, Hh, Hh, Hh, 0);
    for (int r = 0; r < Rr; ++r) {
      hipMemsetAsync(bufB, 0, sizeof(float) * NH, stream);
      scatter_edges<<<Ee / 4, blk, 0, stream>>>(
          h, edge_src + (size_t)r * Ee, edge_dst + (size_t)r * Ee,
          isq + (size_t)r * Nn, bufB);
      // dst-normalize folded into GEMM's per-row A scale; accumulate over r
      gemm_bf16x3<<<gemmGrid, blk, 0, stream>>>(
          bufB, gcn_W + ((size_t)l * Rr + r) * Hh * Hh,
          gcn_b + ((size_t)l * Rr + r) * Hh,
          isq + (size_t)(Rr + r) * Nn,
          bufC, Nn, Hh, Hh, Hh, r > 0 ? 1 : 0);
    }
    hipMemsetAsync(stats, 0, sizeof(float) * 2 * Hh, stream);
    combine_stats<<<statGrid, blk, 0, stream>>>(bufC, hres, 1.f / (float)Rr,
                                                bufB, stats, Nn);
    bn_finalize<<<1, blk, 0, stream>>>(stats, bn_g + (size_t)l * Hh,
                                       bn_b + (size_t)l * Hh, invN, ss);
    bn_apply_act<<<applyGrid, blk, 0, stream>>>(bufB, ss, h, Nn, 0.01f);
  }

  // --- classification head
  gemm_bf16x3<<<gemmGrid, blk, 0, stream>>>(h, W_c1, b_c1, nullptr,
                                            bufB, Nn, Hh, Hh, Hh, 0);
  hipMemsetAsync(stats, 0, sizeof(float) * 2 * Hh, stream);
  col_stats<<<statGrid, blk, 0, stream>>>(bufB, stats, Nn);
  bn_finalize<<<1, blk, 0, stream>>>(stats, g_c, beta_c, invN, ss);
  bn_apply_act<<<applyGrid, blk, 0, stream>>>(bufB, ss, bufC, Nn, 0.f);
  head_gemm<<<(Nn + 255) / 256, blk, 0, stream>>>(bufC, W_c2, b_c2, out, Nn);
}
